// Linear4bit_91001767068469
// MI455X (gfx1250) — 8-bit matrix operand order repaired and checked by reading, compiled, not run on hardware
//
#include <hip/hip_runtime.h>
#include <stdint.h>

typedef __attribute__((ext_vector_type(8))) int v8i;

#define T_DIM  4096
#define K_DIM  4096
#define N_DIM  11008
#define KP     (K_DIM / 2)
#define KSTEPS (K_DIM / 64)
#define LDSTR  80

__global__ __launch_bounds__(256)
void rowsum_u4(const uint32_t* __restrict__ packed, int* __restrict__ out,
               int words_per_row) {
  __shared__ int red[256];
  const uint32_t* p = packed + (size_t)blockIdx.x * words_per_row;
  int s = 0;
  for (int i = threadIdx.x; i < words_per_row; i += 256) {
    uint32_t w = p[i] ^ 0x88888888u;
    uint32_t t = (w & 0x0F0F0F0Fu) + ((w >> 4) & 0x0F0F0F0Fu);
    uint32_t u = (t & 0x00FF00FFu) + ((t >> 8) & 0x00FF00FFu);
    s += (int)((u & 0xFFFFu) + (u >> 16));
  }
  red[threadIdx.x] = s;
  __syncthreads();
  for (int off = 128; off > 0; off >>= 1) {
    if (threadIdx.x < off) red[threadIdx.x] += red[threadIdx.x + off];
    __syncthreads();
  }
  if (threadIdx.x == 0) out[blockIdx.x] = red[0];
}

__device__ __forceinline__ void unpack32(const uint4 r, uint32_t* o) {
  const uint32_t w[4] = {r.x, r.y, r.z, r.w};
#pragma unroll
  for (int j = 0; j < 4; ++j) {
    uint32_t wp = w[j] ^ 0x88888888u;
    uint32_t lo = wp & 0x0F0F0F0Fu;
    uint32_t hi = (wp >> 4) & 0x0F0F0F0Fu;
    o[2 * j]     = __builtin_amdgcn_perm(hi, lo, 0x05010400u);
    o[2 * j + 1] = __builtin_amdgcn_perm(hi, lo, 0x07030602u);
  }
}

__global__ __launch_bounds__(256)
void w4a4_gemm(const uint8_t* __restrict__ pqx,
               const float*   __restrict__ sx,
               const uint8_t* __restrict__ wq,
               const float*   __restrict__ sw,
               const int*     __restrict__ rsX,
               const int*     __restrict__ rsW,
               _Float16*      __restrict__ out) {
  __shared__ __align__(16) uint8_t ldsA[128 * LDSTR];
  __shared__ __align__(16) uint8_t ldsB[128 * LDSTR];

  const int tid  = threadIdx.x;
  const int lane = tid & 31;
  const int wv   = tid >> 5;
  const int wm   = wv >> 2;
  const int wn   = wv & 3;
  const int lh   = lane >> 4;
  const int l15  = lane & 15;

  const int row  = tid >> 1;
  const int half = tid & 1;

  const int mblk = blockIdx.y * 128;
  const int nblk = blockIdx.x * 128;

  const uint8_t* gA = pqx + (size_t)(mblk + row) * KP + half * 16;
  const uint8_t* gB = wq  + (size_t)(nblk + row) * KP + half * 16;

  uint4 ra = *(const uint4*)gA;
  uint4 rb = *(const uint4*)gB;

  v8i acc[4][2];
#pragma unroll
  for (int i = 0; i < 4; ++i)
#pragma unroll
    for (int j = 0; j < 2; ++j)
#pragma unroll
      for (int e = 0; e < 8; ++e) acc[i][j][e] = 0;

  uint8_t* stA = &ldsA[row * LDSTR + half * 32];
  uint8_t* stB = &ldsB[row * LDSTR + half * 32];

  for (int ks = 0; ks < KSTEPS; ++ks) {
    uint32_t o[8];
    unpack32(ra, o);
    ((uint4*)stA)[0] = uint4{o[0], o[1], o[2], o[3]};
    ((uint4*)stA)[1] = uint4{o[4], o[5], o[6], o[7]};
    unpack32(rb, o);
    ((uint4*)stB)[0] = uint4{o[0], o[1], o[2], o[3]};
    ((uint4*)stB)[1] = uint4{o[4], o[5], o[6], o[7]};
    __syncthreads();

    if (ks + 1 < KSTEPS) {
      gA += 32; gB += 32;
      ra = *(const uint4*)gA;
      rb = *(const uint4*)gB;
    }

    v8i bfr[2];
#pragma unroll
    for (int nt = 0; nt < 2; ++nt) {
      const uint8_t* bp = &ldsB[(wn * 32 + nt * 16 + l15) * LDSTR + lh * 16];
      const uint4 q0 = *(const uint4*)(bp);
      const uint4 q1 = *(const uint4*)(bp + 32);
      v8i b;
      b[0] = (int)q0.x; b[1] = (int)q0.y; b[2] = (int)q0.z; b[3] = (int)q0.w;
      b[4] = (int)q1.x; b[5] = (int)q1.y; b[6] = (int)q1.z; b[7] = (int)q1.w;
      bfr[nt] = b;
    }

#pragma unroll
    for (int mt = 0; mt < 4; ++mt) {
      const uint8_t* ap = &ldsA[(wm * 64 + mt * 16 + l15) * LDSTR + lh * 16];
      const uint2 t0 = *(const uint2*)(ap);
      const uint2 t1 = *(const uint2*)(ap + 8);
      const uint2 t2 = *(const uint2*)(ap + 32);
      const uint2 t3 = *(const uint2*)(ap + 40);
      v8i a;
      a[0] = (int)t0.x; a[1] = (int)t0.y; a[2] = (int)t1.x; a[3] = (int)t1.y;
      a[4] = (int)t2.x; a[5] = (int)t2.y; a[6] = (int)t3.x; a[7] = (int)t3.y;

      acc[mt][0] = __builtin_amdgcn_wmma_i32_16x16x64_iu8(
          false, a, false, bfr[0], acc[mt][0], false, false);
      acc[mt][1] = __builtin_amdgcn_wmma_i32_16x16x64_iu8(
          false, a, false, bfr[1], acc[mt][1], false, false);
    }
    __syncthreads();
  }

#pragma unroll
  for (int nt = 0; nt < 2; ++nt) {
    const int   col = nblk + wn * 32 + nt * 16 + l15;
    const float scw = sw[col];
    const int   rbw = rsW[col];
#pragma unroll
    for (int mt = 0; mt < 4; ++mt) {
#pragma unroll
      for (int r = 0; r < 8; ++r) {
        const int   m = mblk + wm * 64 + mt * 16 + lh * 8 + r;
        const int   v = acc[mt][nt][r] - 8 * (rsX[m] + rbw) + 64 * K_DIM;
        const float f = (float)v * sx[m] * scw;
        out[(size_t)m * N_DIM + col] = (_Float16)f;
      }
    }
  }
}

extern "C" void kernel_launch(void* const* d_in, const int* in_sizes, int n_in,
                              void* d_out, int out_size, void* d_ws, size_t ws_size,
                              hipStream_t stream) {
  (void)in_sizes; (void)n_in; (void)out_size; (void)ws_size;

  const uint8_t* pqx = (const uint8_t*)d_in[0];
  const float*   sx  = (const float*)d_in[1];
  const uint8_t* wq  = (const uint8_t*)d_in[2];
  const float*   sw  = (const float*)d_in[3];
  _Float16*      out = (_Float16*)d_out;

  int* rsX = (int*)d_ws;
  int* rsW = rsX + T_DIM;

  rowsum_u4<<<T_DIM, 256, 0, stream>>>((const uint32_t*)pqx, rsX, KP / 4);
  rowsum_u4<<<N_DIM, 256, 0, stream>>>((const uint32_t*)wq,  rsW, KP / 4);

  dim3 grid(N_DIM / 128, T_DIM / 128);
  w4a4_gemm<<<grid, 256, 0, stream>>>(pqx, sx, wq, sw, rsX, rsW, out);
}
